// MultiHeadAttention_12128987644077
// MI455X (gfx1250) — matrix operand order repaired and checked by reading, not compiled, not run on hardware
//
#include <hip/hip_runtime.h>
#include <hip/hip_bf16.h>

// ---------------- problem constants ----------------
#define BATCH 4
#define SEQ   2048
#define EMB   1024
#define HEADS 16
#define HDIM  64

typedef _Float16 half_t;
typedef __attribute__((ext_vector_type(16))) _Float16 v16h;
typedef __attribute__((ext_vector_type(8)))  _Float16 v8h;
typedef __attribute__((ext_vector_type(8)))  float    v8f;
typedef __attribute__((ext_vector_type(4)))  float    v4f;
typedef __attribute__((ext_vector_type(4)))  int      v4i;
typedef __attribute__((address_space(1))) v4i* gv4i_p;   // global int4*
typedef __attribute__((address_space(3))) v4i* sv4i_p;   // LDS int4*

#if defined(__has_builtin)
#if __has_builtin(__builtin_amdgcn_global_load_async_to_lds_b128)
#define HAVE_ASYNC_LDS 1
#endif
#endif

__device__ __forceinline__ void wait_asynccnt0() {
#if defined(__has_builtin) && __has_builtin(__builtin_amdgcn_s_wait_asynccnt)
    __builtin_amdgcn_s_wait_asynccnt(0);
#else
    asm volatile("s_wait_asynccnt 0" ::: "memory");
#endif
}

__device__ __forceinline__ v8f wmma_f16(v16h a, v16h b, v8f c) {
    // D = A(16x32,f16) * B(32x16,f16) + C(16x16,f32)
    return __builtin_amdgcn_wmma_f32_16x16x32_f16(false, a, false, b,
                                                  (short)0, c, false, false);
}

// A-operand (16x32 f16) from a row-major f32 row pointer (row = lane&15).
__device__ __forceinline__ v16h load_A_f32(const float* row, int kk, int hs) {
    const float* p0 = row + kk * 32 + hs * 8;
    const float* p1 = p0 + 16;
    v4f x0 = *(const v4f*)p0, x1 = *(const v4f*)(p0 + 4);
    v4f y0 = *(const v4f*)p1, y1 = *(const v4f*)(p1 + 4);
    v16h a;
#pragma unroll
    for (int j = 0; j < 4; ++j) {
        a[j]      = (_Float16)x0[j];
        a[4 + j]  = (_Float16)x1[j];
        a[8 + j]  = (_Float16)y0[j];
        a[12 + j] = (_Float16)y1[j];
    }
    return a;
}

// B-operand (32x16 f16) from a row-major f32 row pointer (row = column N).
__device__ __forceinline__ v16h load_B_f32(const float* row, int kk, int hs) {
    const float* p = row + kk * 32 + hs * 8;
    v4f x0 = *(const v4f*)p,        x1 = *(const v4f*)(p + 4);
    v4f x2 = *(const v4f*)(p + 16), x3 = *(const v4f*)(p + 20);
    v16h b;
#pragma unroll
    for (int j = 0; j < 4; ++j) {
        b[j]      = (_Float16)x0[j];
        b[4 + j]  = (_Float16)x1[j];
        b[8 + j]  = (_Float16)x2[j];
        b[12 + j] = (_Float16)x3[j];
    }
    return b;
}

// A-operand from row-major f16 (row = lane&15); rowp points at K-tile origin.
__device__ __forceinline__ v16h load_A_f16(const half_t* rowp, int hs) {
    v8h lo = *(const v8h*)(rowp + hs * 8);
    v8h hi = *(const v8h*)(rowp + 16 + hs * 8);
    v16h a;
#pragma unroll
    for (int j = 0; j < 8; ++j) { a[j] = lo[j]; a[8 + j] = hi[j]; }
    return a;
}

// B-operand from row-major f16 (row = column N); rowp points at K-tile origin.
__device__ __forceinline__ v16h load_B_f16(const half_t* rowp, int hs) {
    v8h lo = *(const v8h*)(rowp + hs * 8);
    v8h hi = *(const v8h*)(rowp + 16 + hs * 8);
    v16h b;
#pragma unroll
    for (int j = 0; j < 8; ++j) { b[j] = lo[j]; b[8 + j] = hi[j]; }
    return b;
}

// =====================================================================
// Kernel 1: per-head QKV projection.  out[t,d] = sum_e X[t, h*64+e]*W[d,e] + b[d]
// Wave: 16 tokens x 64 outputs for one head, for Q, K, V.
// Q,K stored row-major f16 [B*H][S][D]; V stored transposed f16 [B*H][D][S].
// =====================================================================
__global__ void k_qkv(const float* __restrict__ value,
                      const float* __restrict__ key,
                      const float* __restrict__ query,
                      const float* __restrict__ Wq, const float* __restrict__ bq,
                      const float* __restrict__ Wk, const float* __restrict__ bk,
                      const float* __restrict__ Wv, const float* __restrict__ bv,
                      half_t* __restrict__ Qh, half_t* __restrict__ Kh,
                      half_t* __restrict__ Vt)
{
    const int lane = threadIdx.x & 31;
    const int wave = threadIdx.x >> 5;
    const int lm   = lane & 15;
    const int hs   = lane >> 4;

    const int wid   = blockIdx.x * 8 + wave;   // 8192 waves
    const int h     = wid & (HEADS - 1);
    const int ttile = wid >> 4;                // 0..511 (tiles of 16 tokens)
    const int tok0  = ttile * 16;              // flat token index into [B*S]
    const int b     = tok0 / SEQ;
    const int s0    = tok0 - b * SEQ;

    const size_t bh = (size_t)b * HEADS + h;
    half_t* outQ = Qh + (bh * SEQ + s0) * HDIM;
    half_t* outK = Kh + (bh * SEQ + s0) * HDIM;
    half_t* outV = Vt + bh * HDIM * SEQ;       // transposed [D][S]

    const float* Xs[3] = {query, key, value};
    const float* Ws[3] = {Wq, Wk, Wv};
    const float* Bs[3] = {bq, bk, bv};

#pragma unroll
    for (int t = 0; t < 3; ++t) {
        const float* Xrow = Xs[t] + ((size_t)tok0 + lm) * EMB + h * HDIM;
        v16h a0 = load_A_f32(Xrow, 0, hs);
        v16h a1 = load_A_f32(Xrow, 1, hs);
#pragma unroll
        for (int nt = 0; nt < 4; ++nt) {
            const int n = nt * 16 + lm;
            const float* Wrow = Ws[t] + n * HDIM;
            v16h b0 = load_B_f32(Wrow, 0, hs);
            v16h b1 = load_B_f32(Wrow, 1, hs);
            const float bias = Bs[t][n];
            v8f c;
#pragma unroll
            for (int v = 0; v < 8; ++v) c[v] = bias;
            c = wmma_f16(a0, b0, c);
            c = wmma_f16(a1, b1, c);
            if (t < 2) {
                half_t* o = (t == 0) ? outQ : outK;
#pragma unroll
                for (int v = 0; v < 8; ++v) {
                    const int m = v + hs * 8;
                    o[(size_t)m * HDIM + n] = (_Float16)c[v];
                }
            } else {
                v8h pk;
#pragma unroll
                for (int v = 0; v < 8; ++v) pk[v] = (_Float16)c[v];
                *(v8h*)(outV + (size_t)n * SEQ + s0 + hs * 8) = pk;
            }
        }
    }
}

// =====================================================================
// Kernel 2: flash attention over one (b,h).  Wave: 16 queries x full D=64.
// All 8 waves of a block share the same (b,h): each 32-wide KV chunk is
// staged into LDS ONCE per block via async load-to-LDS (ASYNCcnt), then
// WMMA B-operands come from LDS (8x less global K/V traffic).
// =====================================================================
#define KPAD 72   // 32 x 72 halves  (row stride 144 B, 16B-aligned, bank-skewed)
#define VPAD 40   // 64 x 40 halves  (row stride  80 B)

__global__ void k_attn(const half_t* __restrict__ Qh,
                       const half_t* __restrict__ Kh,
                       const half_t* __restrict__ Vt,
                       const int* __restrict__ mask,
                       half_t* __restrict__ O)
{
    __shared__ half_t Kl[32][KPAD];
    __shared__ half_t Vl[64][VPAD];
    __shared__ half_t pstage[8][16 * 32];  // per-wave C->A staging of P

    const int tid  = threadIdx.x;
    const int lane = tid & 31;
    const int wave = tid >> 5;
    const int lm   = lane & 15;
    const int hs   = lane >> 4;

    const int bh = blockIdx.y;             // 0..63
    const int b  = bh >> 4;
    const int h  = bh & 15;
    const int qt = blockIdx.x * 8 + wave;  // 0..127
    const int q0 = qt * 16;

    const half_t* Qbase = Qh + ((size_t)bh * SEQ + q0) * HDIM;
    const half_t* Kbase = Kh + (size_t)bh * SEQ * HDIM;
    const half_t* Vbase = Vt + (size_t)bh * HDIM * SEQ;
    const int*    mrow  = mask + (size_t)b * SEQ * SEQ;
    half_t* myP = &pstage[wave][0];

    // per-thread cooperative staging assignment (256 threads)
    const int krow = tid >> 3, kseg = tid & 7;   // K: 32 rows x 8 segs of 8 halves
    const int vrow = tid >> 2, vseg = tid & 3;   // V: 64 rows x 4 segs of 8 halves

    // Q A-tiles (reused across the whole KV loop)
    const half_t* qr = Qbase + (size_t)lm * HDIM;
    v16h qa0 = load_A_f16(qr, hs);
    v16h qa1 = load_A_f16(qr + 32, hs);

    v8f acc[4];
#pragma unroll
    for (int nt = 0; nt < 4; ++nt)
#pragma unroll
        for (int v = 0; v < 8; ++v) acc[nt][v] = 0.0f;

    float rowmax[8], lsum[8];
#pragma unroll
    for (int v = 0; v < 8; ++v) { rowmax[v] = -__builtin_inff(); lsum[v] = 0.0f; }

    for (int kv0 = 0; kv0 < SEQ; kv0 += 32) {
        __syncthreads();  // previous chunk fully consumed before overwrite

        const half_t* kg = Kbase + (size_t)(kv0 + krow) * HDIM + kseg * 8;
        const half_t* vg = Vbase + (size_t)vrow * SEQ + kv0 + vseg * 8;
#ifdef HAVE_ASYNC_LDS
        __builtin_amdgcn_global_load_async_to_lds_b128(
            (gv4i_p)kg, (sv4i_p)&Kl[krow][kseg * 8], 0, 0);
        __builtin_amdgcn_global_load_async_to_lds_b128(
            (gv4i_p)vg, (sv4i_p)&Vl[vrow][vseg * 8], 0, 0);
        wait_asynccnt0();
#else
        *(v8h*)&Kl[krow][kseg * 8] = *(const v8h*)kg;
        *(v8h*)&Vl[vrow][vseg * 8] = *(const v8h*)vg;
#endif
        if (kv0 + 32 < SEQ) {  // prefetch next chunk (global_prefetch_b8)
            __builtin_prefetch(kg + 32 * HDIM, 0, 1);
            __builtin_prefetch(vg + 32, 0, 1);
        }
        __syncthreads();

        // ---- energy tiles (2 x 16 kv columns) from LDS, scaled + masked ----
        v8f e[2];
#pragma unroll
        for (int kvt = 0; kvt < 2; ++kvt) {
            const half_t* Krow = &Kl[kvt * 16 + lm][0];
            v16h kb0 = load_B_f16(Krow, hs);
            v16h kb1 = load_B_f16(Krow + 32, hs);
            v8f c;
#pragma unroll
            for (int v = 0; v < 8; ++v) c[v] = 0.0f;
            c = wmma_f16(qa0, kb0, c);
            c = wmma_f16(qa1, kb1, c);
            const int kvb = kv0 + kvt * 16;
#pragma unroll
            for (int v = 0; v < 8; ++v) {
                const int qrow = q0 + v + hs * 8;
                const int mv = mrow[(size_t)qrow * SEQ + kvb + lm];
                const float ev = c[v] * 0.125f;  // 1/sqrt(64)
                e[kvt][v] = (mv == 0) ? -__builtin_inff() : ev;
            }
        }

        // ---- online softmax update (row stats per VGPR index) ----
        float p[2][8], scalev[8];
#pragma unroll
        for (int v = 0; v < 8; ++v) {
            float cm = fmaxf(e[0][v], e[1][v]);
#pragma unroll
            for (int sh = 1; sh < 16; sh <<= 1) cm = fmaxf(cm, __shfl_xor(cm, sh));
            const float nmax = fmaxf(rowmax[v], cm);
            const float nm   = fmaxf(nmax, -3.0e38f);   // avoid inf-inf
            const float sc   = __expf(rowmax[v] - nm);
            rowmax[v] = nmax;
            p[0][v] = __expf(e[0][v] - nm);
            p[1][v] = __expf(e[1][v] - nm);
            float ps = p[0][v] + p[1][v];
#pragma unroll
            for (int sh = 1; sh < 16; sh <<= 1) ps += __shfl_xor(ps, sh);
            lsum[v] = lsum[v] * sc + ps;
            scalev[v] = sc;
        }
#pragma unroll
        for (int nt = 0; nt < 4; ++nt)
#pragma unroll
            for (int v = 0; v < 8; ++v) acc[nt][v] *= scalev[v];

        // ---- P: C-layout -> A-layout via per-wave LDS tile ----
#pragma unroll
        for (int kvt = 0; kvt < 2; ++kvt)
#pragma unroll
            for (int v = 0; v < 8; ++v)
                myP[(v + hs * 8) * 32 + kvt * 16 + lm] = (_Float16)p[kvt][v];
        // same-wave DS ops are in-order: safe to read back without a barrier
        v16h pa;
        {
            const half_t* rp = myP + lm * 32 + hs * 8;
#pragma unroll
            for (int j = 0; j < 8; ++j) { pa[j] = rp[j]; pa[8 + j] = rp[16 + j]; }
        }

        // ---- O += P * V  (B from transposed V tile in LDS) ----
#pragma unroll
        for (int nt = 0; nt < 4; ++nt) {
            const half_t* vr = &Vl[nt * 16 + lm][hs * 8];
            v8h l0 = *(const v8h*)vr;
            v8h l1 = *(const v8h*)(vr + 16);
            v16h vb;
#pragma unroll
            for (int j = 0; j < 8; ++j) { vb[j] = l0[j]; vb[8 + j] = l1[j]; }
            acc[nt] = wmma_f16(pa, vb, acc[nt]);
        }
    }

    // ---- finalize: divide by row sum, store head-concat f16 ----
    float inv[8];
#pragma unroll
    for (int v = 0; v < 8; ++v) inv[v] = (lsum[v] > 0.0f) ? (1.0f / lsum[v]) : 0.0f;
#pragma unroll
    for (int nt = 0; nt < 4; ++nt)
#pragma unroll
        for (int v = 0; v < 8; ++v) {
            const size_t tok = (size_t)b * SEQ + q0 + v + hs * 8;
            O[tok * EMB + h * HDIM + nt * 16 + lm] = (_Float16)(acc[nt][v] * inv[v]);
        }
}

// =====================================================================
// Kernel 3a: Wfc f32 -> f16 (2 MB, lives in L2 for the GEMM)
// =====================================================================
__global__ void k_cvt(const float* __restrict__ W, half_t* __restrict__ W16, int n)
{
    const int i = blockIdx.x * blockDim.x + threadIdx.x;
    if (i < n) W16[i] = (_Float16)W[i];
}

// =====================================================================
// Kernel 3b: FC GEMM  out[m,n] = sum_k X[m,k]*Wfc[n,k] + bfc[n]
// M = B*S = 8192, N = K = 1024.  Wave: 16 tokens x 64 columns.
// =====================================================================
__global__ void k_fc(const half_t* __restrict__ X,
                     const half_t* __restrict__ W16,
                     const float* __restrict__ bfc,
                     float* __restrict__ out)
{
    const int lane = threadIdx.x & 31;
    const int wave = threadIdx.x >> 5;
    const int lm   = lane & 15;
    const int hs   = lane >> 4;

    const int mt = blockIdx.x;                       // 0..511
    const int n0 = blockIdx.y * 512 + wave * 64;     // 0..960

    const half_t* Arow = X + ((size_t)mt * 16 + lm) * EMB;

    v8f acc[4];
#pragma unroll
    for (int nt = 0; nt < 4; ++nt) {
        const float bias = bfc[n0 + nt * 16 + lm];
#pragma unroll
        for (int v = 0; v < 8; ++v) acc[nt][v] = bias;
    }

    for (int k0 = 0; k0 < EMB; k0 += 32) {
        v16h a = load_A_f16(Arow + k0, hs);
#pragma unroll
        for (int nt = 0; nt < 4; ++nt) {
            const half_t* Wrow = W16 + (size_t)(n0 + nt * 16 + lm) * EMB + k0;
            v16h bm = load_B_f16(Wrow, hs);
            acc[nt] = wmma_f16(a, bm, acc[nt]);
        }
    }

#pragma unroll
    for (int nt = 0; nt < 4; ++nt)
#pragma unroll
        for (int v = 0; v < 8; ++v)
            out[((size_t)mt * 16 + v + hs * 8) * EMB + n0 + nt * 16 + lm] = acc[nt][v];
}

// =====================================================================
extern "C" void kernel_launch(void* const* d_in, const int* in_sizes, int n_in,
                              void* d_out, int out_size, void* d_ws, size_t ws_size,
                              hipStream_t stream)
{
    const float* value = (const float*)d_in[0];
    const float* key   = (const float*)d_in[1];
    const float* query = (const float*)d_in[2];
    const int*   mask  = (const int*)d_in[3];
    const float* Wq  = (const float*)d_in[4];
    const float* bq  = (const float*)d_in[5];
    const float* Wk  = (const float*)d_in[6];
    const float* bk  = (const float*)d_in[7];
    const float* Wv  = (const float*)d_in[8];
    const float* bv  = (const float*)d_in[9];
    const float* Wfc = (const float*)d_in[10];
    const float* bfc = (const float*)d_in[11];
    float* out = (float*)d_out;

    char* ws = (char*)d_ws;
    half_t* Qh  = (half_t*)(ws);
    half_t* Kh  = (half_t*)(ws + ((size_t)16 << 20));
    half_t* Vt  = (half_t*)(ws + ((size_t)32 << 20));
    half_t* Oa  = (half_t*)(ws + ((size_t)48 << 20));
    half_t* W16 = (half_t*)(ws + ((size_t)64 << 20));

    // 1) QKV projections: 8192 waves (512 token-tiles x 16 heads), 8 waves/block
    k_qkv<<<dim3(1024), 256, 0, stream>>>(value, key, query,
                                          Wq, bq, Wk, bk, Wv, bv,
                                          Qh, Kh, Vt);
    // 2) flash attention: grid (S/128, B*H)
    k_attn<<<dim3(16, 64), 256, 0, stream>>>(Qh, Kh, Vt, mask, Oa);
    // 3) weight convert + FC GEMM
    k_cvt<<<dim3((EMB * EMB) / 256), 256, 0, stream>>>(Wfc, W16, EMB * EMB);
    k_fc<<<dim3(512, 2), 256, 0, stream>>>(Oa, W16, bfc, out);
}
